// Attention_29850022707772
// MI455X (gfx1250) — matrix operand order repaired and checked by reading, not compiled, not run on hardware
//
#include <hip/hip_runtime.h>
#include <hip/hip_bf16.h>

// ---------------------------------------------------------------------------
// Fused attention block for MI455X (gfx1250, wave32, WMMA + TDM).
//   x[4,2048,512] -> LN -> QKV GEMM -> RoPE -> 8-head attention -> out GEMM
// Matrix math: v_wmma_f32_16x16x32_f16. K/V and GEMM-A tiles staged into LDS
// via tensor_load_to_lds (TENSORcnt), double-buffered in the attention loop.
// ---------------------------------------------------------------------------

typedef __attribute__((ext_vector_type(16))) _Float16 v16h;
typedef __attribute__((ext_vector_type(8)))  _Float16 v8h;
typedef __attribute__((ext_vector_type(4)))  _Float16 v4h;
typedef __attribute__((ext_vector_type(8)))  float    v8f;
typedef __attribute__((ext_vector_type(4)))  unsigned int u32x4;
typedef __attribute__((ext_vector_type(8)))  int          i32x8;
typedef __attribute__((ext_vector_type(4)))  int          i32x4;

#define BATCH   4
#define SEQ     2048
#define DIM     512
#define HEADS   8
#define DHEAD   64
#define INNER   512           // HEADS*DHEAD
#define QKVN    1536          // 3*INNER
#define ROWS    8192          // BATCH*SEQ
#define LN_EPS  1e-5f

#if defined(__AMDGCN__) && __has_builtin(__builtin_amdgcn_tensor_load_to_lds)
#define HAVE_TDM 1
#warning "gfx1250 device pass: TDM (tensor_load_to_lds) path ENABLED"
#else
#define HAVE_TDM 0
#if defined(__AMDGCN__)
#warning "gfx1250 device pass: TDM builtin MISSING -> cooperative-copy fallback"
#endif
#endif

__device__ __forceinline__ v8f zero8() {
  v8f z;
#pragma unroll
  for (int j = 0; j < 8; ++j) z[j] = 0.0f;
  return z;
}

// A fragment: 16x32 f16 tile, row-major, leading dim ld (elems).
// lane: row = lane&15 ; K = koff..+15, koff=(lane>>4)*16
__device__ __forceinline__ v16h load_a_frag(const _Float16* base, int ld) {
  const int l = threadIdx.x & 31;
  const _Float16* p = base + (size_t)(l & 15) * ld + ((l >> 4) << 4);
  v8h lo = *(const v8h*)p;
  v8h hi = *(const v8h*)(p + 8);
  v16h a;
#pragma unroll
  for (int j = 0; j < 8; ++j) { a[j] = lo[j]; a[j + 8] = hi[j]; }
  return a;
}

// B fragment: 32x16 tile given as B^T row-major ([N][K], leading dim ld).
// lane: col N = lane&15 ; K = (lane>>4)*16 .. +15 (contiguous 32B)
__device__ __forceinline__ v16h load_b_frag(const _Float16* baseT, int ld) {
  const int l = threadIdx.x & 31;
  const _Float16* p = baseT + (size_t)(l & 15) * ld + ((l >> 4) << 4);
  return *(const v16h*)p;
}

__device__ __forceinline__ v8f wmma_f16(v16h a, v16h b, v8f c) {
  return __builtin_amdgcn_wmma_f32_16x16x32_f16(false, a, false, b,
                                                (short)0, c, false, false);
}

#if HAVE_TDM
__device__ __forceinline__ unsigned lds_off(const void* p) {
  return (unsigned)(unsigned long long)p;   // flat shared addr: low 32 = LDS offset
}
// 2D TDM load, f16 elements, LDS row padding enabled.
// pad_i: code for DWORDs-per-row (4 -> 32 DW = 64 f16, 7 -> 256 DW = 512 f16)
// pad_a: code for pad DWORDs (3 -> 4 DW = 16B = 8 f16)
__device__ __forceinline__ void tdm_load_2d(unsigned ldsoff, const void* gsrc,
                                            unsigned td0, unsigned td1,
                                            unsigned stride0, unsigned tile0,
                                            unsigned tile1, unsigned pad_i,
                                            unsigned pad_a) {
  unsigned long long ga = (unsigned long long)gsrc;
  u32x4 g0;
  g0[0] = 1u;                                     // count=1, user descriptor
  g0[1] = ldsoff;                                 // lds_addr (bytes)
  g0[2] = (unsigned)(ga & 0xFFFFFFFFu);           // global_addr lo
  g0[3] = (unsigned)((ga >> 32) & 0x1FFFFFFu) | (2u << 30);  // ga hi | type=2
  i32x8 g1;
  g1[0] = (int)((1u << 16) | (1u << 20) | (pad_i << 22) | (pad_a << 25));
  g1[1] = (int)((td0 & 0xFFFFu) << 16);           // tensor_dim0 lo16
  g1[2] = (int)((td0 >> 16) | ((td1 & 0xFFFFu) << 16));
  g1[3] = (int)((td1 >> 16) | (tile0 << 16));     // tile_dim0
  g1[4] = (int)tile1;                             // tile_dim1 (tile_dim2=0)
  g1[5] = (int)stride0;                           // tensor_dim0_stride lo32
  g1[6] = 0;
  g1[7] = 0;
  i32x4 z4 = {0, 0, 0, 0};
#if defined(__clang_major__) && __clang_major__ >= 23
  i32x8 z8 = {0, 0, 0, 0, 0, 0, 0, 0};
  __builtin_amdgcn_tensor_load_to_lds(g0, g1, z4, z4, z8, 0);
#else
  __builtin_amdgcn_tensor_load_to_lds(g0, g1, z4, z4, 0);
#endif
}
#endif  // HAVE_TDM

// ---------------------------------------------------------------------------
// K1: transpose + cast f32 [K][N] -> f16 [N][K]  (weights -> B^T layout)
// ---------------------------------------------------------------------------
__global__ __launch_bounds__(256) void tcast_kernel(
    const float* __restrict__ src, _Float16* __restrict__ dst, int K, int N) {
  int idx = blockIdx.x * 256 + threadIdx.x;
  if (idx >= K * N) return;
  int k = idx & (K - 1);          // K = 512
  int n = idx >> 9;
  dst[(size_t)n * K + k] = (_Float16)src[(size_t)k * N + n];
}

// ---------------------------------------------------------------------------
// K2: LayerNorm, one wave per row of 512, write f16 xn
// ---------------------------------------------------------------------------
__global__ __launch_bounds__(256) void ln_kernel(
    const float* __restrict__ x, const float* __restrict__ gamma,
    const float* __restrict__ beta, _Float16* __restrict__ xn) {
  const int wave = threadIdx.x >> 5, lane = threadIdx.x & 31;
  const int row  = blockIdx.x * 8 + wave;
  const float* xr = x + (size_t)row * DIM;
  float v[16];
  float s = 0.f, s2 = 0.f;
#pragma unroll
  for (int j = 0; j < 4; ++j) {
    float4 t = *(const float4*)(xr + j * 128 + lane * 4);
    v[j*4+0] = t.x; v[j*4+1] = t.y; v[j*4+2] = t.z; v[j*4+3] = t.w;
    s  += t.x + t.y + t.z + t.w;
    s2 += t.x*t.x + t.y*t.y + t.z*t.z + t.w*t.w;
  }
#pragma unroll
  for (int m = 1; m < 32; m <<= 1) {
    s  += __shfl_xor(s,  m, 32);
    s2 += __shfl_xor(s2, m, 32);
  }
  const float mean = s * (1.0f / DIM);
  const float var  = s2 * (1.0f / DIM) - mean * mean;
  const float rstd = rsqrtf(var + LN_EPS);
  _Float16* orow = xn + (size_t)row * DIM;
#pragma unroll
  for (int j = 0; j < 4; ++j) {
    int col = j * 128 + lane * 4;
    v4h o;
#pragma unroll
    for (int i = 0; i < 4; ++i)
      o[i] = (_Float16)((v[j*4+i] - mean) * rstd * gamma[col+i] + beta[col+i]);
    *(v4h*)(orow + col) = o;
  }
}

// ---------------------------------------------------------------------------
// Shared GEMM body: A tile (16 x 512) staged in LDS (pitch 520), B from
// global. Fully unrolled, software-pipelined k-loop (static cur/next regs).
// ---------------------------------------------------------------------------
#define A_PITCH 520

__device__ __forceinline__ void gemm_tile_16x64(
    const _Float16* Alds, const _Float16* BT, int n0, v8f acc[4]) {
  v16h a[2];
  v16h b[2][4];
  a[0] = load_a_frag(Alds, A_PITCH);
#pragma unroll
  for (int t = 0; t < 4; ++t)
    b[0][t] = load_b_frag(BT + (size_t)(n0 + t * 16) * DIM, DIM);
#pragma unroll
  for (int i = 0; i < 16; ++i) {            // DIM/32 steps, fully unrolled
    const int cur = i & 1, nxt = cur ^ 1;
    if (i < 15) {                           // issue next-stage loads first
      const int kn = (i + 1) * 32;
      a[nxt] = load_a_frag(Alds + kn, A_PITCH);
#pragma unroll
      for (int t = 0; t < 4; ++t)
        b[nxt][t] = load_b_frag(BT + (size_t)(n0 + t * 16) * DIM + kn, DIM);
    }
#pragma unroll
    for (int t = 0; t < 4; ++t) acc[t] = wmma_f16(a[cur], b[cur][t], acc[t]);
  }
}

__device__ __forceinline__ void stage_a_tile(_Float16* Atile,
                                             const _Float16* Arow) {
#if HAVE_TDM
  if ((threadIdx.x >> 5) == 0) {    // one wave issues the DMA (wave-uniform)
    tdm_load_2d(lds_off(Atile), Arow, DIM, 16, DIM, DIM, 16, /*pad_i=*/7, /*pad_a=*/3);
    __builtin_amdgcn_s_wait_tensorcnt(0);
  }
#else
  for (int i = threadIdx.x; i < 16 * (DIM / 8); i += 256) {
    int r = i >> 6, c = (i & 63) * 8;
    *(v8h*)&Atile[r * A_PITCH + c] = *(const v8h*)(Arow + (size_t)r * DIM + c);
  }
#endif
  __syncthreads();
}

// K3: QKV GEMM  xn[8192,512] @ wqkv -> qkv f16 [8192][1536]; grid (512,3)x256
__global__ __launch_bounds__(256) void gemm_qkv_kernel(
    const _Float16* __restrict__ A, const _Float16* __restrict__ BT,
    _Float16* __restrict__ C) {
  __shared__ __align__(16) _Float16 Atile[16 * A_PITCH];
  const int wave = threadIdx.x >> 5, lane = threadIdx.x & 31;
  const int m0 = blockIdx.x * 16;
  const int n0 = (blockIdx.y * 8 + wave) * 64;
  stage_a_tile(Atile, A + (size_t)m0 * DIM);
  v8f acc[4] = {zero8(), zero8(), zero8(), zero8()};
  gemm_tile_16x64(Atile, BT, n0, acc);
  const int cr = lane & 15, rh = (lane >> 4) * 8;
#pragma unroll
  for (int t = 0; t < 4; ++t)
#pragma unroll
    for (int r = 0; r < 8; ++r)
      C[(size_t)(m0 + rh + r) * QKVN + n0 + t * 16 + cr] = (_Float16)acc[t][r];
}

// K6: out GEMM  ctx[8192,512] @ w_out + b_out -> f32; grid 512 x 256
__global__ __launch_bounds__(256) void gemm_out_kernel(
    const _Float16* __restrict__ A, const _Float16* __restrict__ BT,
    const float* __restrict__ bias, float* __restrict__ out) {
  __shared__ __align__(16) _Float16 Atile[16 * A_PITCH];
  const int wave = threadIdx.x >> 5, lane = threadIdx.x & 31;
  const int m0 = blockIdx.x * 16;
  const int n0 = wave * 64;
  stage_a_tile(Atile, A + (size_t)m0 * INNER);
  v8f acc[4] = {zero8(), zero8(), zero8(), zero8()};
  gemm_tile_16x64(Atile, BT, n0, acc);
  const int cr = lane & 15, rh = (lane >> 4) * 8;
#pragma unroll
  for (int t = 0; t < 4; ++t)
#pragma unroll
    for (int r = 0; r < 8; ++r) {
      int col = n0 + t * 16 + cr;
      // final output: written once, never re-read -> non-temporal store
      __builtin_nontemporal_store(acc[t][r] + bias[col],
                                  out + (size_t)(m0 + rh + r) * DIM + col);
    }
}

// ---------------------------------------------------------------------------
// K4: RoPE + head split.  Q,K -> [bh][n][64] (roped), V -> V^T [bh][64][n]
// ---------------------------------------------------------------------------
__global__ __launch_bounds__(256) void rope_kernel(
    const _Float16* __restrict__ qkv, _Float16* __restrict__ Q,
    _Float16* __restrict__ K, _Float16* __restrict__ VT) {
  int idx = blockIdx.x * 256 + threadIdx.x;         // 32*2048*64 total
  const int dh = idx & 63;
  const int nn = (idx >> 6) & (SEQ - 1);
  const int bh = idx >> 17;
  const int h = bh & 7, bb = bh >> 3;
  const _Float16* qr = qkv + ((size_t)bb * SEQ + nn) * QKVN;
  const float inv = exp2f(-(float)dh * (13.287712379549449f / 64.0f)); // log2(1e4)
  float sa, ca;
  __sincosf((float)nn * inv, &sa, &ca);
  const int dp = (dh + 63) & 63;
  const float qv = (float)qr[h * 64 + dh],         qp = (float)qr[h * 64 + dp];
  const float kv = (float)qr[INNER + h * 64 + dh], kp = (float)qr[INNER + h * 64 + dp];
  const size_t ho = ((size_t)bh * SEQ + nn) * DHEAD + dh;
  Q[ho] = (_Float16)(qv * ca + qp * sa);
  K[ho] = (_Float16)(kv * ca + kp * sa);
  VT[((size_t)bh * DHEAD + dh) * SEQ + nn] = qr[2 * INNER + h * 64 + dh];
}

// ---------------------------------------------------------------------------
// K5: flash attention. grid (32 bh, 32 qblk), block 128 (4 waves).
// K/V 64x64 blocks staged in LDS via TDM, double-buffered (TENSORcnt).
// ---------------------------------------------------------------------------
#define KV_PITCH 72

__global__ __launch_bounds__(128) void attn_kernel(
    const _Float16* __restrict__ Q, const _Float16* __restrict__ Km,
    const _Float16* __restrict__ VT, _Float16* __restrict__ ctx) {
  __shared__ __align__(16) _Float16 kbuf[2][64 * KV_PITCH];
  __shared__ __align__(16) _Float16 vbuf[2][64 * KV_PITCH];
  __shared__ __align__(16) _Float16 Plds[4][16 * KV_PITCH];
  const int wave = threadIdx.x >> 5, lane = threadIdx.x & 31;
  const int bh = blockIdx.x;
  const int q0 = blockIdx.y * 64 + wave * 16;
  const _Float16* Qh = Q  + (size_t)bh * SEQ * DHEAD;
  const _Float16* Kh = Km + (size_t)bh * SEQ * DHEAD;
  const _Float16* Vh = VT + (size_t)bh * DHEAD * SEQ;

  const v16h qf0 = load_a_frag(Qh + (size_t)q0 * DHEAD +  0, DHEAD);
  const v16h qf1 = load_a_frag(Qh + (size_t)q0 * DHEAD + 32, DHEAD);

  v8f o[4] = {zero8(), zero8(), zero8(), zero8()};
  float m[8], lsum[8];
#pragma unroll
  for (int r = 0; r < 8; ++r) { m[r] = -1.0e30f; lsum[r] = 0.0f; }

  _Float16* Pw = Plds[wave];
  const int cr = lane & 15, rh = (lane >> 4) * 8;
  const float scale = 0.125f;                       // 1/sqrt(64)

#if HAVE_TDM
  if (wave == 0) {                                  // prologue: stage block 0
    tdm_load_2d(lds_off(kbuf[0]), Kh, DHEAD, SEQ, DHEAD, DHEAD, 64, 4, 3);
    tdm_load_2d(lds_off(vbuf[0]), Vh, SEQ, DHEAD, SEQ, 64, DHEAD, 4, 3);
  }
#endif

#pragma unroll 2                                    // make buffer parity static
  for (int kb = 0, it = 0; kb < SEQ; kb += 64, ++it) {
#if HAVE_TDM
    const int p = it & 1;
    __syncthreads();                                // prev reads of buf[1-p] done
    if (wave == 0) {
      if (kb + 64 < SEQ) {                          // prefetch next block (DMA)
        const _Float16* kn = Kh + (size_t)(kb + 64) * DHEAD;
        const _Float16* vn = Vh + (size_t)(kb + 64);
        tdm_load_2d(lds_off(kbuf[1 - p]), kn, DHEAD, SEQ, DHEAD, DHEAD, 64, 4, 3);
        tdm_load_2d(lds_off(vbuf[1 - p]), vn, SEQ, DHEAD, SEQ, 64, DHEAD, 4, 3);
        __builtin_amdgcn_s_wait_tensorcnt(2);       // current block complete
      } else {
        __builtin_amdgcn_s_wait_tensorcnt(0);
      }
    }
    __syncthreads();                                // buffer published
#else
    const int p = 0;
    __syncthreads();
    for (int i = threadIdx.x; i < 64 * 8; i += 128) {
      int r = i >> 3, c = (i & 7) * 8;
      *(v8h*)&kbuf[0][r * KV_PITCH + c] =
          *(const v8h*)(Kh + (size_t)(kb + r) * DHEAD + c);
      *(v8h*)&vbuf[0][r * KV_PITCH + c] =
          *(const v8h*)(Vh + (size_t)r * SEQ + kb + c);
    }
    __syncthreads();
#endif
    const _Float16* kt = kbuf[p];
    const _Float16* vt = vbuf[p];

    // ---- S = scale * Q K^T  (16 x 64) ----
    v8f s[4];
#pragma unroll
    for (int t = 0; t < 4; ++t) {
      const _Float16* kbase = kt + (t * 16) * KV_PITCH;
      v16h b0 = load_b_frag(kbase +  0, KV_PITCH);
      v16h b1 = load_b_frag(kbase + 32, KV_PITCH);
      v8f ss = zero8();
      ss = wmma_f16(qf0, b0, ss);
      ss = wmma_f16(qf1, b1, ss);
#pragma unroll
      for (int r = 0; r < 8; ++r) ss[r] *= scale;
      s[t] = ss;
    }
    // ---- online softmax (rows live in 16-lane column groups) ----
#pragma unroll
    for (int r = 0; r < 8; ++r) {
      float mx = fmaxf(fmaxf(s[0][r], s[1][r]), fmaxf(s[2][r], s[3][r]));
#pragma unroll
      for (int msk = 1; msk < 16; msk <<= 1) mx = fmaxf(mx, __shfl_xor(mx, msk, 32));
      const float mnew = fmaxf(m[r], mx);
      const float corr = __expf(m[r] - mnew);
      m[r] = mnew;
      float rs = 0.f;
#pragma unroll
      for (int t = 0; t < 4; ++t) {
        float pv = __expf(s[t][r] - mnew);
        s[t][r] = pv;
        rs += pv;
      }
#pragma unroll
      for (int msk = 1; msk < 16; msk <<= 1) rs += __shfl_xor(rs, msk, 32);
      lsum[r] = lsum[r] * corr + rs;
#pragma unroll
      for (int t = 0; t < 4; ++t) o[t][r] *= corr;
    }
    // ---- stage P (C/D layout -> row-major LDS -> A fragments) ----
#pragma unroll
    for (int t = 0; t < 4; ++t)
#pragma unroll
      for (int r = 0; r < 8; ++r)
        Pw[(rh + r) * KV_PITCH + t * 16 + cr] = (_Float16)s[t][r];
    asm volatile("s_wait_dscnt 0" ::: "memory");    // LDS RAW within wave
    v16h pf0 = load_a_frag(Pw +  0, KV_PITCH);
    v16h pf1 = load_a_frag(Pw + 32, KV_PITCH);
    // ---- O += P V ----
#pragma unroll
    for (int d = 0; d < 4; ++d) {
      const _Float16* vb = vt + (d * 16) * KV_PITCH;
      v16h b0 = load_b_frag(vb +  0, KV_PITCH);
      v16h b1 = load_b_frag(vb + 32, KV_PITCH);
      o[d] = wmma_f16(pf0, b0, o[d]);
      o[d] = wmma_f16(pf1, b1, o[d]);
    }
  }
  // ---- normalize + store ctx f16 [row][h*64 + d] ----
  const int b = bh >> 3, h = bh & 7;
#pragma unroll
  for (int t = 0; t < 4; ++t)
#pragma unroll
    for (int r = 0; r < 8; ++r)
      ctx[((size_t)b * SEQ + q0 + rh + r) * INNER + h * 64 + t * 16 + cr] =
          (_Float16)(o[t][r] * (1.0f / lsum[r]));
}

// ---------------------------------------------------------------------------
extern "C" void kernel_launch(void* const* d_in, const int* in_sizes, int n_in,
                              void* d_out, int out_size, void* d_ws, size_t ws_size,
                              hipStream_t stream) {
  (void)in_sizes; (void)n_in; (void)out_size; (void)ws_size;
  const float* x        = (const float*)d_in[0];
  const float* ln_gamma = (const float*)d_in[1];
  const float* ln_beta  = (const float*)d_in[2];
  const float* w_qkv    = (const float*)d_in[3];
  const float* w_out    = (const float*)d_in[4];
  const float* b_out    = (const float*)d_in[5];
  float* out = (float*)d_out;

  // workspace layout (256B aligned)
  char* ws = (char*)d_ws;
  size_t off = 0;
  auto take = [&](size_t bytes) -> char* {
    char* p = ws + off;
    off += (bytes + 255) & ~(size_t)255;
    return p;
  };
  _Float16* wqkvT = (_Float16*)take((size_t)QKVN * DIM * 2);   // [1536][512]
  _Float16* woutT = (_Float16*)take((size_t)DIM * INNER * 2);  // [512][512]
  _Float16* xn    = (_Float16*)take((size_t)ROWS * DIM * 2);   // [8192][512]
  _Float16* qkv   = (_Float16*)take((size_t)ROWS * QKVN * 2);  // [8192][1536]
  _Float16* Qr    = (_Float16*)take((size_t)32 * SEQ * DHEAD * 2);
  _Float16* Kr    = (_Float16*)take((size_t)32 * SEQ * DHEAD * 2);
  _Float16* VT    = (_Float16*)take((size_t)32 * DHEAD * SEQ * 2);
  _Float16* ctx   = xn;  // xn dead after QKV GEMM -> reuse

  tcast_kernel<<<(DIM * QKVN) / 256, 256, 0, stream>>>(w_qkv, wqkvT, DIM, QKVN);
  tcast_kernel<<<(INNER * DIM) / 256, 256, 0, stream>>>(w_out, woutT, INNER, DIM);
  ln_kernel<<<ROWS / 8, 256, 0, stream>>>(x, ln_gamma, ln_beta, xn);
  gemm_qkv_kernel<<<dim3(ROWS / 16, QKVN / (64 * 8)), 256, 0, stream>>>(xn, wqkvT, qkv);
  rope_kernel<<<(32 * SEQ * DHEAD) / 256, 256, 0, stream>>>(qkv, Qr, Kr, VT);
  attn_kernel<<<dim3(32, SEQ / 64), 128, 0, stream>>>(Qr, Kr, VT, ctx);
  gemm_out_kernel<<<ROWS / 16, 256, 0, stream>>>(ctx, woutT, b_out, out);
}
